// MFFM_50989851738866
// MI455X (gfx1250) — matrix operand order repaired and checked by reading, not compiled, not run on hardware
//
#include <hip/hip_runtime.h>
#include <hip/hip_bf16.h>
#include <math.h>

// Problem constants
#define B_   4
#define C_   256
#define HW_  4096
#define L_   4096

typedef __attribute__((ext_vector_type(16))) __bf16 v16bf;
typedef __attribute__((ext_vector_type(8)))  float  v8f;
typedef unsigned int u32x4 __attribute__((ext_vector_type(4)));
typedef int          i32x8 __attribute__((ext_vector_type(8)));
typedef int          i32x4 __attribute__((ext_vector_type(4)));

#ifndef __has_builtin
#define __has_builtin(x) 0
#endif
#if __has_builtin(__builtin_amdgcn_tensor_load_to_lds) && \
    __has_builtin(__builtin_amdgcn_s_wait_tensorcnt)
#define HAVE_TDM 1
#else
#define HAVE_TDM 0
#endif

__device__ __forceinline__ unsigned short f2bf(float f) {
  union { float f; unsigned int u; } c; c.f = f;
  unsigned int u = c.u;
  return (unsigned short)((u + 0x7FFFu + ((u >> 16) & 1u)) >> 16);
}
__device__ __forceinline__ float sigm(float x) { return 1.0f / (1.0f + expf(-x)); }

// ---------------------------------------------------------------------------
// WMMA 16x16x32 bf16 tile.  A in LDS [rows x ldA] bf16, B in LDS [16 x ldB]
// (row = N, so K pairs are contiguous -> dword LDS loads for both fragments).
// Fragment layouts per CDNA5 ISA 7.12.2.
// ---------------------------------------------------------------------------
__device__ __forceinline__ v8f wmma_tile(const unsigned short* lA, int mBase, int ldA, int kA,
                                         const unsigned short* lB, int ldB, int kB, v8f acc) {
  int lane = threadIdx.x & 31;
  int mh = lane >> 4, ml = lane & 15;
  union { v16bf v; unsigned int u[8]; } a, b;
#pragma unroll
  for (int i = 0; i < 8; ++i) {
    int e = 2 * i;
    int ka = kA + 16 * (e >> 3) + 8 * mh + (e & 7);           // A: K coord of elem e
    a.u[i] = *(const unsigned int*)(lA + (mBase + ml) * ldA + ka);
    int kb = kB + 16 * (e >> 3) + 8 * mh + (e & 7);           // B: K coord of elem e
    b.u[i] = *(const unsigned int*)(lB + ml * ldB + kb);
  }
  return __builtin_amdgcn_wmma_f32_16x16x32_bf16(false, a.v, false, b.v,
                                                 (short)0, acc, false, false);
}

#if HAVE_TDM
// ---------------------------------------------------------------------------
// Tensor Data Mover: async-load a 2D bf16 tile (tile_h rows x tile_w elems,
// row stride = stride elems) from global into LDS, compacted row-major.
// D# layout per CDNA5 ISA.
// ---------------------------------------------------------------------------
__device__ __forceinline__ void tdm_load_2d(unsigned lds_off, const void* gptr,
                                            unsigned stride, unsigned tile_w,
                                            unsigned tile_h, unsigned tensor_w,
                                            unsigned tensor_h) {
  unsigned long long ga = (unsigned long long)(size_t)gptr;    // 57-bit global addr
  u32x4 g0;
  g0[0] = 1u;                                                  // count=1 valid user D#
  g0[1] = lds_off;                                             // lds_addr [63:32]
  g0[2] = (unsigned)(ga & 0xFFFFFFFFu);                        // global_addr lo
  g0[3] = (unsigned)((ga >> 32) & 0x01FFFFFFu) | (2u << 30);   // addr hi | type=2
  i32x8 g1;
  g1[0] = (int)(1u << 16);                                     // data_size=1 (2 bytes)
  g1[1] = (int)(tensor_w << 16);                               // tensor_dim0 lo16
  g1[2] = (int)((tensor_w >> 16) | (tensor_h << 16));          // dim0 hi | dim1 lo
  g1[3] = (int)((tensor_h >> 16) | (tile_w << 16));            // dim1 hi | tile_dim0
  g1[4] = (int)tile_h;                                         // tile_dim1 (dim2=0)
  g1[5] = (int)stride;                                         // tensor_dim0_stride lo32
  g1[6] = 0;
  g1[7] = 0;
  i32x4 z4 = {0, 0, 0, 0};
#if defined(__clang_major__) && (__clang_major__ >= 23)
  i32x8 z8 = {0, 0, 0, 0, 0, 0, 0, 0};
  __builtin_amdgcn_tensor_load_to_lds(g0, g1, z4, z4, z8, 0);
#else
  __builtin_amdgcn_tensor_load_to_lds(g0, g1, z4, z4, 0);
#endif
}
#endif

// ---------------------------------------------------------------------------
// Attention pooling
// ---------------------------------------------------------------------------
__global__ void k_pool(const float* __restrict__ R, const float* __restrict__ Dx,
                       float* poolW, float* poolH, float* poolC) {
  int c = blockIdx.x, b = blockIdx.y, img = blockIdx.z;
  const float* src = (img ? Dx : R) + ((size_t)(b * C_ + c)) * HW_;
  int t = threadIdx.x;  // 64 threads
  float colsum = 0.f;
  for (int h = 0; h < 64; ++h) colsum += src[h * 64 + t];
  float rowsum = 0.f;
  for (int w = 0; w < 64; ++w) rowsum += src[t * 64 + w];
  size_t base = ((size_t)(img * B_ + b) * C_ + c) * 64;
  poolW[base + t] = colsum * (1.f / 64.f);
  poolH[base + t] = rowsum * (1.f / 64.f);
  __shared__ float red[64];
  red[t] = colsum;
  __syncthreads();
  if (t == 0) {
    float s = 0.f;
    for (int i = 0; i < 64; ++i) s += red[i];
    poolC[(size_t)(img * B_ + b) * C_ + c] = s * (1.f / 4096.f);
  }
}

struct GateP { const float *c1w, *c1b, *bng, *bnb, *bnm, *bnv, *c2w, *c2b; };
struct GatesArgs { GateP g[4]; };

__global__ void k_gates(GatesArgs ga, const float* poolW, const float* poolH,
                        float* gateW, float* gateH) {
  int img = blockIdx.x, b = blockIdx.y, br = blockIdx.z;
  GateP gp = ga.g[img * 2 + br];
  const float* pool = (br == 0 ? poolW : poolH) + ((size_t)(img * B_ + b) * C_) * 64;
  float* gout = (br == 0 ? gateW : gateH) + ((size_t)(img * B_ + b) * C_) * 64;
  __shared__ float z[8 * 64];
  int tid = threadIdx.x;  // 256
  for (int idx = tid; idx < 8 * 64; idx += 256) {
    int m = idx >> 6, w = idx & 63;
    float s = gp.c1b[m];
    for (int c = 0; c < C_; ++c) s += gp.c1w[m * C_ + c] * pool[c * 64 + w];
    s = (s - gp.bnm[m]) * rsqrtf(gp.bnv[m] + 1e-5f) * gp.bng[m] + gp.bnb[m];
    float cl = fminf(fmaxf(s + 3.f, 0.f), 6.f);
    z[idx] = s * cl * (1.f / 6.f);
  }
  __syncthreads();
  for (int idx = tid; idx < C_ * 64; idx += 256) {
    int cc = idx >> 6, w = idx & 63;
    float s = gp.c2b[cc];
    for (int m = 0; m < 8; ++m) s += gp.c2w[cc * 8 + m] * z[m * 64 + w];
    gout[idx] = sigm(s);
  }
}

__global__ void k_cgate(const float* fc1r, const float* fc2r,
                        const float* fc1d, const float* fc2d,
                        const float* poolC, float* gateC) {
  int img = blockIdx.x, b = blockIdx.y;
  const float* fc1 = img ? fc1d : fc1r;
  const float* fc2 = img ? fc2d : fc2r;
  const float* pc = poolC + (size_t)(img * B_ + b) * C_;
  float* gc = gateC + (size_t)(img * B_ + b) * C_;
  __shared__ float zz[8];
  int tid = threadIdx.x;  // 64
  if (tid < 8) {
    float s = 0.f;
    for (int c = 0; c < C_; ++c) s += fc1[tid * C_ + c] * pc[c];
    zz[tid] = fmaxf(s, 0.f);
  }
  __syncthreads();
  for (int cc = tid; cc < C_; cc += 64) {
    float s = 0.f;
    for (int m = 0; m < 8; ++m) s += fc2[cc * 8 + m] * zz[m];
    gc[cc] = sigm(s);
  }
}

__global__ void k_apply(const float* __restrict__ R, const float* __restrict__ Dx,
                        const float* gateW, const float* gateH, const float* gateC,
                        float* attR, float* attD) {
  int img = blockIdx.z;
  size_t idx = (size_t)blockIdx.x * 256 + threadIdx.x;
  const float* src = img ? Dx : R;
  float* dst = img ? attD : attR;
  int b = idx >> 20, c = (idx >> 12) & 255, p = idx & 4095;
  int w = p & 63, h = p >> 6;
  size_t g = (size_t)(img * B_ + b) * C_ + c;
  dst[idx] = src[idx] * gateW[g * 64 + w] * gateH[g * 64 + h] * gateC[g];
}

__global__ void k_max(const float* attR, const float* attD, float* mbuf) {
  int i = blockIdx.x * 256 + threadIdx.x;
  int b = i >> 12, p = i & 4095;
  float m = -3.4e38f;
  for (int c = 0; c < C_; ++c) {
    m = fmaxf(m, attR[((size_t)(b * C_ + c)) * HW_ + p]);
    m = fmaxf(m, attD[((size_t)(b * C_ + c)) * HW_ + p]);
  }
  mbuf[i] = m;
}

__global__ void k_sa(const float* mbuf, const float* sa7, const float* sa1, float* sabuf) {
  int i = blockIdx.x * 256 + threadIdx.x;
  int b = i >> 12, p = i & 4095, h = p >> 6, w = p & 63;
  float acc = 0.f;
  for (int dy = 0; dy < 7; ++dy)
    for (int dx = 0; dx < 7; ++dx) {
      int yy = h + dy - 3, xx = w + dx - 3;
      if (yy >= 0 && yy < 64 && xx >= 0 && xx < 64)
        acc += sa7[dy * 7 + dx] * mbuf[(b << 12) + yy * 64 + xx];
    }
  sabuf[i] = sigm(acc) * sa1[0];
}

__global__ void k_outr(const float* attR, const float* sabuf, const float* R, float* outr) {
  size_t idx = (size_t)blockIdx.x * 256 + threadIdx.x;
  int b = idx >> 20, p = idx & 4095;
  outr[idx] = attR[idx] * sabuf[(b << 12) + p] + R[idx];
}

// ---------------------------------------------------------------------------
// GEMM1: out0[b,o,p] = sa[b,p]*(conv_w[o,:] . cat(attR,attD)[b,:,p]) + conv_b[o]
// M=64, K=512, chunked (B computed on the fly from f32 sources).
// ---------------------------------------------------------------------------
__global__ void k_gemm_conv(const float* W, const float* bias,
                            const float* attR, const float* attD,
                            const float* sabuf, float* out0) {
  __shared__ __align__(16) unsigned short lA[64 * 32];
  __shared__ __align__(16) unsigned short lB[16 * 32];
  int tid = threadIdx.x, b = blockIdx.y, n0 = blockIdx.x * 16;
  v8f acc = {};
  for (int k0 = 0; k0 < 512; k0 += 32) {
    __syncthreads();
    for (int i = tid; i < 64 * 32; i += 128) {
      int m = i >> 5, kk = (i & 31) + k0;
      lA[i] = f2bf(W[m * 512 + kk]);
    }
    for (int i = tid; i < 512; i += 128) {
      int n = i >> 5, kk = (i & 31) + k0;
      float v = (kk < 256) ? attR[((size_t)(b * C_ + kk)) * HW_ + n0 + n]
                           : attD[((size_t)(b * C_ + kk - 256)) * HW_ + n0 + n];
      lB[i] = f2bf(v);
    }
    __syncthreads();
    acc = wmma_tile(lA, (tid >> 5) * 16, 32, 0, lB, 32, 0, acc);
  }
  int lane = tid & 31, wave = tid >> 5, mh = lane >> 4, ml = lane & 15;
#pragma unroll
  for (int r = 0; r < 8; ++r) {
    int m = wave * 16 + r + 8 * mh, n = n0 + ml;
    out0[((size_t)(b * 64 + m)) * HW_ + n] = acc[r] * sabuf[(b << 12) + n] + bias[m];
  }
}

// LayerNorm over 64 channels per token -> bf16 token-major
__global__ void k_ln(const float* out0, const float* lng, const float* lnb,
                     unsigned short* xnbf) {
  int j = blockIdx.x * 64 + threadIdx.x;  // 16384 tokens
  int b = j >> 12, t = j & 4095;
  size_t base = (size_t)b * 64 * HW_ + t;
  float s = 0.f, s2 = 0.f;
  for (int c = 0; c < 64; ++c) {
    float v = out0[base + (size_t)c * HW_];
    s += v; s2 += v * v;
  }
  float mu = s * (1.f / 64.f);
  float var = s2 * (1.f / 64.f) - mu * mu;
  float inv = rsqrtf(var + 1e-5f);
  for (int c = 0; c < 64; ++c) {
    float v = out0[base + (size_t)c * HW_];
    xnbf[(size_t)j * 64 + c] = f2bf((v - mu) * inv * lng[c] + lnb[c]);
  }
}

// GEMM2: xz[b,t,i] = in_w[i,:] . xn[b,t,:]   M=256, K=64.  A staged once,
// B tile (16 tokens x 64) fetched by the Tensor Data Mover.
__global__ void k_gemm_inproj(const float* inw, const unsigned short* xnbf, float* xz) {
  __shared__ __align__(16) unsigned short lA[256 * 64];  // 32 KB
  __shared__ __align__(16) unsigned short lB[16 * 64];   // 2 KB
  int tid = threadIdx.x, b = blockIdx.y, n0 = blockIdx.x * 16;
#if HAVE_TDM
  if (tid < 32) {
    tdm_load_2d((unsigned)(size_t)(void*)lB,
                xnbf + ((size_t)(b * L_) + n0) * 64,
                64, 64, 16, 64, (unsigned)(B_ * L_));
  }
#else
  for (int i = tid; i < 16 * 64; i += 128)
    lB[i] = xnbf[((size_t)(b * L_) + n0 + (i >> 6)) * 64 + (i & 63)];
#endif
  for (int i = tid; i < 256 * 64; i += 128) lA[i] = f2bf(inw[i]);
#if HAVE_TDM
  if (tid < 32) __builtin_amdgcn_s_wait_tensorcnt(0);
#endif
  __syncthreads();
  v8f accs[4];
  v8f zero = {};
#pragma unroll
  for (int j = 0; j < 4; ++j) accs[j] = zero;
#pragma unroll
  for (int k0 = 0; k0 < 64; k0 += 32)
#pragma unroll
    for (int j = 0; j < 4; ++j)
      accs[j] = wmma_tile(lA, ((tid >> 5) + 4 * j) * 16, 64, k0, lB, 64, k0, accs[j]);
  int lane = tid & 31, wave = tid >> 5, mh = lane >> 4, ml = lane & 15;
#pragma unroll
  for (int j = 0; j < 4; ++j)
#pragma unroll
    for (int r = 0; r < 8; ++r) {
      int m = (wave + 4 * j) * 16 + r + 8 * mh, n = n0 + ml;
      xz[((size_t)(b * L_) + n) * 256 + m] = accs[j][r];
    }
}

// depthwise causal conv (k=4) over L + SiLU
__global__ void k_conv(const float* xz, const float* cw, const float* cb, float* ubuf) {
  size_t idx = (size_t)blockIdx.x * 256 + threadIdx.x;
  int tok = idx >> 7, d = idx & 127;
  int b = tok >> 12, t = tok & 4095;
  float acc = cb[d];
#pragma unroll
  for (int k = 0; k < 4; ++k) {
    int tt = t - 3 + k;
    if (tt >= 0) acc += cw[d * 4 + k] * xz[((size_t)(b << 12) + tt) * 256 + d];
  }
  ubuf[idx] = acc * sigm(acc);
}

__global__ void k_xproj(const float* ubuf, const float* xw, float* dbl) {
  size_t idx = (size_t)blockIdx.x * 256 + threadIdx.x;
  int tok = idx / 36, jj = idx % 36;
  float acc = 0.f;
  for (int d = 0; d < 128; ++d) acc += xw[jj * 128 + d] * ubuf[(size_t)tok * 128 + d];
  dbl[idx] = acc;
}

__global__ void k_delta(const float* dbl, const float* dtw, const float* dtb, float* delta) {
  size_t idx = (size_t)blockIdx.x * 256 + threadIdx.x;
  int tok = idx >> 7, d = idx & 127;
  float s = dtb[d];
#pragma unroll
  for (int r = 0; r < 4; ++r) s += dtw[d * 4 + r] * dbl[(size_t)tok * 36 + r];
  delta[idx] = (s > 20.f) ? s : log1pf(expf(s));
}

// selective scan: 8192 independent first-order recurrences; s-reduction via
// shfl; prefetch 64 tokens ahead to hide HBM latency on the dependent chain.
__global__ void k_scan(const float* delta, const float* ubuf, const float* dbl,
                       const float* A_log, float* ysb) {
  int dg = blockIdx.x, b = blockIdx.y;
  int tid = threadIdx.x;  // 256
  int dl = tid >> 4, s = tid & 15;
  int d = dg * 16 + dl;
  float A = -expf(A_log[d * 16 + s]);
  float h = 0.f;
  for (int t = 0; t < L_; ++t) {
    size_t tok = ((size_t)b << 12) + t;
    if ((t & 63) == 0 && t + 64 < L_) {
      size_t tp = tok + 64;
      __builtin_prefetch(&delta[tp * 128 + d], 0, 1);
      __builtin_prefetch(&ubuf[tp * 128 + d], 0, 1);
      __builtin_prefetch(&dbl[tp * 36 + s], 0, 1);
    }
    float dv = delta[tok * 128 + d];
    float uv = ubuf[tok * 128 + d];
    float Bv = dbl[tok * 36 + 4 + s];
    float Cv = dbl[tok * 36 + 20 + s];
    h = expf(dv * A) * h + dv * Bv * uv;
    float p = h * Cv;
    p += __shfl_xor(p, 1);
    p += __shfl_xor(p, 2);
    p += __shfl_xor(p, 4);
    p += __shfl_xor(p, 8);
    if (s == 0) ysb[tok * 128 + d] = p;
  }
}

__global__ void k_gate(const float* ysb, const float* ubuf, const float* Dp,
                       const float* xz, unsigned short* ybf) {
  size_t idx = (size_t)blockIdx.x * 256 + threadIdx.x;
  int tok = idx >> 7, d = idx & 127;
  float z = xz[(size_t)tok * 256 + 128 + d];
  float y = (ysb[idx] + ubuf[idx] * Dp[d]) * (z * sigm(z));
  ybf[idx] = f2bf(y);
}

// GEMM4: out_proj [64x128] + exact GELU -> br2.  A staged once, B via TDM.
__global__ void k_gemm_outproj(const float* ow, const unsigned short* ybf, float* br2) {
  __shared__ __align__(16) unsigned short lA[64 * 128];  // 16 KB
  __shared__ __align__(16) unsigned short lB[16 * 128];  // 4 KB
  int tid = threadIdx.x, b = blockIdx.y, n0 = blockIdx.x * 16;
#if HAVE_TDM
  if (tid < 32) {
    tdm_load_2d((unsigned)(size_t)(void*)lB,
                ybf + ((size_t)(b * L_) + n0) * 128,
                128, 128, 16, 128, (unsigned)(B_ * L_));
  }
#else
  for (int i = tid; i < 16 * 128; i += 128)
    lB[i] = ybf[((size_t)(b * L_) + n0 + (i >> 7)) * 128 + (i & 127)];
#endif
  for (int i = tid; i < 64 * 128; i += 128) lA[i] = f2bf(ow[i]);
#if HAVE_TDM
  if (tid < 32) __builtin_amdgcn_s_wait_tensorcnt(0);
#endif
  __syncthreads();
  v8f acc = {};
#pragma unroll
  for (int k0 = 0; k0 < 128; k0 += 32)
    acc = wmma_tile(lA, (tid >> 5) * 16, 128, k0, lB, 128, k0, acc);
  int lane = tid & 31, wave = tid >> 5, mh = lane >> 4, ml = lane & 15;
#pragma unroll
  for (int r = 0; r < 8; ++r) {
    int m = wave * 16 + r + 8 * mh, n = n0 + ml;
    float v = acc[r];
    v = 0.5f * v * (1.f + erff(v * 0.70710678f));   // exact GELU
    br2[((size_t)(b * 64 + m)) * HW_ + n] = v;
  }
}

// GEMM5: end conv 192->64; B rows 0-63 = out0, 64-127 = relu(bn(out0)) on the
// fly, 128-191 = br2.  A (64x192) staged once; B chunked.
__global__ void k_gemm_end(const float* ew, const float* eb,
                           const float* out0, const float* br2,
                           const float* bg, const float* bb,
                           const float* bm, const float* bv, float* outp) {
  __shared__ __align__(16) unsigned short lA[64 * 192];  // 24 KB
  __shared__ __align__(16) unsigned short lB[16 * 32];
  int tid = threadIdx.x, b = blockIdx.y, n0 = blockIdx.x * 16;
  for (int i = tid; i < 64 * 192; i += 128) lA[i] = f2bf(ew[i]);
  v8f acc = {};
  for (int k0 = 0; k0 < 192; k0 += 32) {
    __syncthreads();
    for (int i = tid; i < 512; i += 128) {
      int n = i >> 5, kk = (i & 31) + k0;
      int t = n0 + n;
      float v;
      if (kk < 64) {
        v = out0[((size_t)(b * 64 + kk)) * HW_ + t];
      } else if (kk < 128) {
        int c = kk - 64;
        float f = out0[((size_t)(b * 64 + c)) * HW_ + t];
        v = fmaxf(0.f, (f - bm[c]) * rsqrtf(bv[c] + 1e-5f) * bg[c] + bb[c]);
      } else {
        v = br2[((size_t)(b * 64 + kk - 128)) * HW_ + t];
      }
      lB[i] = f2bf(v);
    }
    __syncthreads();
    acc = wmma_tile(lA, (tid >> 5) * 16, 192, k0, lB, 32, 0, acc);
  }
  int lane = tid & 31, wave = tid >> 5, mh = lane >> 4, ml = lane & 15;
#pragma unroll
  for (int r = 0; r < 8; ++r) {
    int m = wave * 16 + r + 8 * mh, n = n0 + ml;
    outp[((size_t)(b * 64 + m)) * HW_ + n] = acc[r] + eb[m];
  }
}

// ---------------------------------------------------------------------------
extern "C" void kernel_launch(void* const* d_in, const int* in_sizes, int n_in,
                              void* d_out, int out_size, void* d_ws, size_t ws_size,
                              hipStream_t stream) {
  (void)in_sizes; (void)n_in; (void)out_size; (void)ws_size;
  const float** in = (const float**)d_in;
  // insertion-order flattening of setup_inputs(): r, d, params{...}
  const float* R = in[0];
  const float* Dx = in[1];
  const float* sa7 = in[38];
  const float* sa1 = in[39];
  const float* conv_w = in[40];
  const float* conv_b = in[41];
  const float *bng = in[42], *bnb = in[43], *bnm = in[44], *bnv = in[45];
  const float *m_lng = in[46], *m_lnb = in[47], *m_inw = in[48];
  const float *m_cw = in[49], *m_cb = in[50], *m_xw = in[51];
  const float *m_dtw = in[52], *m_dtb = in[53], *m_Alog = in[54];
  const float *m_D = in[55], *m_ow = in[56];
  const float *end_w = in[57], *end_b = in[58];

  float* outr = (float*)d_out;                       // [4,256,64,64]
  float* outp = outr + (size_t)B_ * C_ * HW_;        // [4,64,64,64]

  size_t off = 0;
  char* ws = (char*)d_ws;
  auto AF = [&](size_t n) { float* p = (float*)(ws + off); off += n * 4; return p; };
  auto AU = [&](size_t n) { unsigned short* p = (unsigned short*)(ws + off); off += n * 2; return p; };
  float* poolW = AF(2 * B_ * C_ * 64);
  float* poolH = AF(2 * B_ * C_ * 64);
  float* poolC = AF(2 * B_ * C_);
  float* gateW = AF(2 * B_ * C_ * 64);
  float* gateH = AF(2 * B_ * C_ * 64);
  float* gateC = AF(2 * B_ * C_);
  float* attR  = AF((size_t)B_ * C_ * HW_);
  float* attD  = AF((size_t)B_ * C_ * HW_);
  float* mbuf  = AF(B_ * HW_);
  float* sabuf = AF(B_ * HW_);
  float* out0  = AF((size_t)B_ * 64 * HW_);
  unsigned short* xnbf = AU((size_t)B_ * L_ * 64);
  float* xz    = AF((size_t)B_ * L_ * 256);
  float* ubuf  = AF((size_t)B_ * L_ * 128);
  float* dbl   = AF((size_t)B_ * L_ * 36);
  float* delta = AF((size_t)B_ * L_ * 128);
  float* ysb   = AF((size_t)B_ * L_ * 128);
  unsigned short* ybf = AU((size_t)B_ * L_ * 128);
  float* br2   = AF((size_t)B_ * 64 * HW_);

  GatesArgs ga;
  ga.g[0] = { in[10], in[11], in[12], in[13], in[14], in[15], in[16], in[17] }; // r,W
  ga.g[1] = { in[2],  in[3],  in[4],  in[5],  in[6],  in[7],  in[8],  in[9]  }; // r,H
  ga.g[2] = { in[28], in[29], in[30], in[31], in[32], in[33], in[34], in[35] }; // d,W
  ga.g[3] = { in[20], in[21], in[22], in[23], in[24], in[25], in[26], in[27] }; // d,H

  k_pool<<<dim3(C_, B_, 2), 64, 0, stream>>>(R, Dx, poolW, poolH, poolC);
  k_gates<<<dim3(2, B_, 2), 256, 0, stream>>>(ga, poolW, poolH, gateW, gateH);
  k_cgate<<<dim3(2, B_), 64, 0, stream>>>(in[18], in[19], in[36], in[37], poolC, gateC);
  k_apply<<<dim3(16384, 1, 2), 256, 0, stream>>>(R, Dx, gateW, gateH, gateC, attR, attD);
  k_max<<<dim3(64), 256, 0, stream>>>(attR, attD, mbuf);
  k_sa<<<dim3(64), 256, 0, stream>>>(mbuf, sa7, sa1, sabuf);
  k_outr<<<dim3(16384), 256, 0, stream>>>(attR, sabuf, R, outr);
  k_gemm_conv<<<dim3(256, B_), 128, 0, stream>>>(conv_w, conv_b, attR, attD, sabuf, out0);
  k_ln<<<dim3(256), 64, 0, stream>>>(out0, m_lng, m_lnb, xnbf);
  k_gemm_inproj<<<dim3(256, B_), 128, 0, stream>>>(m_inw, xnbf, xz);
  k_conv<<<dim3(8192), 256, 0, stream>>>(xz, m_cw, m_cb, ubuf);
  k_xproj<<<dim3(2304), 256, 0, stream>>>(ubuf, m_xw, dbl);
  k_delta<<<dim3(8192), 256, 0, stream>>>(dbl, m_dtw, m_dtb, delta);
  k_scan<<<dim3(8, B_), 256, 0, stream>>>(delta, ubuf, dbl, m_Alog, ysb);
  k_gate<<<dim3(8192), 256, 0, stream>>>(ysb, ubuf, m_D, xz, ybf);
  k_gemm_outproj<<<dim3(256, B_), 128, 0, stream>>>(m_ow, ybf, br2);
  k_gemm_end<<<dim3(256, B_), 128, 0, stream>>>(end_w, end_b, out0, br2,
                                                bng, bnb, bnm, bnv, outp);
}
